// SCT_Memory_25494925869411
// MI455X (gfx1250) — matrix operand order repaired and checked by reading, not compiled, not run on hardware
//
#include <hip/hip_runtime.h>

typedef __attribute__((ext_vector_type(16))) __bf16   v16bf;
typedef __attribute__((ext_vector_type(8)))  float    v8f;
typedef __attribute__((ext_vector_type(4)))  float    v4f;
typedef __attribute__((ext_vector_type(4)))  unsigned v4u;
typedef __attribute__((ext_vector_type(2)))  unsigned v2u;

union Frag { v4u u[2]; v16bf v; };   // 16 bf16 = 8 VGPRs

__device__ __forceinline__ unsigned as_u(float f) {
    union { float f; unsigned u; } c; c.f = f; return c.u;
}
__device__ __forceinline__ float as_f(unsigned u) {
    union { unsigned u; float f; } c; c.u = u; return c.f;
}
// f32 -> (bf16 hi bits, bf16 lo bits); hi is RNE, lo is truncated (error ~2^-16 rel)
__device__ __forceinline__ void split_bf16(float x, unsigned& hi, unsigned& lo) {
    unsigned u = as_u(x);
    unsigned r = u + 0x7FFFu + ((u >> 16) & 1u);
    hi = r >> 16;
    float hf = as_f(r & 0xFFFF0000u);          // value of hi as f32
    lo = as_u(x - hf) >> 16;
}

// ---------------- kernel 1: split inputs[256*256] f32 -> bf16 hi/lo planes ----
__global__ void split_inputs(const float* __restrict__ A,
                             unsigned short* __restrict__ hi,
                             unsigned short* __restrict__ lo, int n)
{
    int i = blockIdx.x * blockDim.x + threadIdx.x;
    if (i < n) {
        unsigned h, l;
        split_bf16(A[i], h, l);
        hi[i] = (unsigned short)h;
        lo[i] = (unsigned short)l;
    }
}

// ---------------- kernel 2: bf16x3 WMMA GEMM --------------------------------
// out[256, N] = (inputs @ features^T) / TEMP
// Workgroup: 320 threads = 10 waves = 2 Mpos x 5 Npos; wave block = 32x32 (2x2 tiles).
// WG tile: 64 M x 160 N. grid = (N/160, 256/64) = (625, 4).
#define D     256
#define MT    64
#define NT    160
// LDS row layout: [hi 0..255 | pad 8 | lo 0..255 | pad 16] = 536 shorts (1072 B).
// 1072 B = 268 dwords == 12 (mod 64) -> 16 rows map to 16 distinct 4-bank groups.
#define RS    536
#define LO    264

__global__ __launch_bounds__(320) void sct_mem_scores(
    const unsigned short* __restrict__ Ahi,   // [256, 256] bf16 bits, row-major
    const unsigned short* __restrict__ Alo,
    const float*          __restrict__ F,     // features [N, 256] f32
    float*                __restrict__ out,   // [256, N] f32
    int N)
{
    __shared__ unsigned short Ash[MT * RS];   //  68.6 KB (hi|lo interleaved per row)
    __shared__ unsigned short Bsh[NT * RS];   // 171.5 KB

    const int tid    = threadIdx.x;
    const int lane   = tid & 31;
    const int wave   = tid >> 5;
    const int half   = lane >> 4;
    const int l16    = lane & 15;
    const int n_base = blockIdx.x * NT;
    const int m_base = blockIdx.y * MT;

    // ---- stage A slab (already bf16; pure copy, uint4-coalesced) ----
    {
        const unsigned short* planes[2] = { Ahi, Alo };
        // per plane: MT rows x 32 uint4 (8 shorts each) = 2048 -> 4096 total
        for (int i = tid; i < 2 * MT * (D / 8); i += 320) {
            int p   = i >> 11;            // / (MT*32)
            int r   = (i >> 5) & (MT - 1);
            int c8  = i & 31;
            v4u v = *(const v4u*)(planes[p] + (size_t)(m_base + r) * D + c8 * 8);
            *(v4u*)&Ash[r * RS + p * LO + c8 * 8] = v;
        }
    }
    // ---- stage + split B slab (f32 -> bf16 hi/lo), float4-coalesced ----
    {
        for (int i = tid; i < NT * (D / 4); i += 320) {
            int r  = i >> 6;              // / (D/4)
            int c4 = i & 63;
            v4f v = *(const v4f*)(F + (size_t)(n_base + r) * D + c4 * 4);
            unsigned h0, h1, h2, h3, l0, l1, l2, l3;
            split_bf16(v[0], h0, l0);
            split_bf16(v[1], h1, l1);
            split_bf16(v[2], h2, l2);
            split_bf16(v[3], h3, l3);
            v2u hp = { h0 | (h1 << 16), h2 | (h3 << 16) };
            v2u lp = { l0 | (l1 << 16), l2 | (l3 << 16) };
            *(v2u*)&Bsh[r * RS + c4 * 4]      = hp;
            *(v2u*)&Bsh[r * RS + LO + c4 * 4] = lp;
        }
    }
    __syncthreads();

    const int mi = wave / 5;      // 0..1  -> M offset mi*32
    const int ni = wave % 5;      // 0..4  -> N offset ni*32

    // Hoisted integer base indices (NOT pointers -> keeps addrspace(3) ds loads).
    // All loop offsets from these bases fit the 16-bit DS immediate.
    int rA[2], rB[2];
#pragma unroll
    for (int t = 0; t < 2; ++t)
        rA[t] = (mi * 32 + t * 16 + l16) * RS + 8 * half;
#pragma unroll
    for (int u = 0; u < 2; ++u)
        rB[u] = (ni * 32 + u * 16 + l16) * RS + 8 * half;

    v8f acc[2][2] = {};

    // unroll capped at 2: live fragments ~16 (128 VGPRs) + acc 32 -> no spills
#pragma unroll 2
    for (int kc = 0; kc < D; kc += 32) {
        Frag ah[2], al[2], bh[2], bl[2];
#pragma unroll
        for (int t = 0; t < 2; ++t) {
            // A 16x32 bf16 fragment: i<8 -> k = kc+8*half+i ; i>=8 -> +16
            ah[t].u[0] = *(const v4u*)&Ash[rA[t] + kc];
            ah[t].u[1] = *(const v4u*)&Ash[rA[t] + kc + 16];
            al[t].u[0] = *(const v4u*)&Ash[rA[t] + LO + kc];
            al[t].u[1] = *(const v4u*)&Ash[rA[t] + LO + kc + 16];
        }
#pragma unroll
        for (int u = 0; u < 2; ++u) {
            // B 32x16 bf16 fragment: i<8 -> k = kc+8*half+i ; i>=8 -> +16
            bh[u].u[0] = *(const v4u*)&Bsh[rB[u] + kc];
            bh[u].u[1] = *(const v4u*)&Bsh[rB[u] + kc + 16];
            bl[u].u[0] = *(const v4u*)&Bsh[rB[u] + LO + kc];
            bl[u].u[1] = *(const v4u*)&Bsh[rB[u] + LO + kc + 16];
        }
#pragma unroll
        for (int t = 0; t < 2; ++t)
#pragma unroll
            for (int u = 0; u < 2; ++u) {
                acc[t][u] = __builtin_amdgcn_wmma_f32_16x16x32_bf16(
                    false, ah[t].v, false, bh[u].v, (short)0, acc[t][u], false, false);
                acc[t][u] = __builtin_amdgcn_wmma_f32_16x16x32_bf16(
                    false, ah[t].v, false, bl[u].v, (short)0, acc[t][u], false, false);
                acc[t][u] = __builtin_amdgcn_wmma_f32_16x16x32_bf16(
                    false, al[t].v, false, bh[u].v, (short)0, acc[t][u], false, false);
            }
    }

    // ---- scale by 1/TEMP and store; C/D layout: row m = v + 8*half, col n = l16
    const float inv_temp = 1.0f / 0.07f;
#pragma unroll
    for (int t = 0; t < 2; ++t)
#pragma unroll
        for (int u = 0; u < 2; ++u) {
            float* op = out
                + (size_t)(m_base + mi * 32 + t * 16 + half * 8) * N
                + (n_base + ni * 32 + u * 16 + l16);
#pragma unroll
            for (int v = 0; v < 8; ++v)
                op[(size_t)v * N] = acc[t][u][v] * inv_temp;
        }
}

extern "C" void kernel_launch(void* const* d_in, const int* in_sizes, int n_in,
                              void* d_out, int out_size, void* d_ws, size_t ws_size,
                              hipStream_t stream) {
    const float* inputs   = (const float*)d_in[0];   // [B, D] f32
    const float* features = (const float*)d_in[2];   // [N, D] f32
    float* out = (float*)d_out;                      // [B, N] f32

    const int Bsz = in_sizes[1];          // 256
    const int Dd  = in_sizes[0] / Bsz;    // 256 (kernel specialized for 256)
    const int N   = in_sizes[2] / Dd;     // 100000

    unsigned short* Ahi = (unsigned short*)d_ws;            // 128 KB
    unsigned short* Alo = Ahi + (size_t)Bsz * Dd;           // 128 KB

    const int nA = Bsz * Dd;
    split_inputs<<<dim3((nA + 255) / 256), dim3(256), 0, stream>>>(inputs, Ahi, Alo, nA);

    dim3 grid(N / NT, Bsz / MT);          // (625, 4)
    dim3 block(320);                      // 10 waves
    sct_mem_scores<<<grid, block, 0, stream>>>(Ahi, Alo, features, out, N);
}
